// LinearAttention_27144193311318
// MI455X (gfx1250) — matrix operand order repaired and checked by reading, not compiled, not run on hardware
//
#include <hip/hip_runtime.h>
#include <hip/hip_bf16.h>

// MI455X (gfx1250) implementation of the linear-attention block.
// Dominant cost: x @ qk_w^T GEMM (131072 x 96 x 192) -> done with
// v_wmma_f32_16x16x32_bf16. All reductions are fixed-order (deterministic).

typedef __attribute__((ext_vector_type(16))) __bf16 bf16x16;
typedef __attribute__((ext_vector_type(8)))  float  floatx8;

#define BATCH 8
#define NTOK  16384
#define CH    96
#define QKC   192
#define NH    4
#define HD    24
#define HW    128

// ---------------------------------------------------------------- K0: W->bf16
__global__ void la_wconv(const float* __restrict__ w, __bf16* __restrict__ wb) {
    int i = blockIdx.x * blockDim.x + threadIdx.x;
    if (i < QKC * CH) wb[i] = (__bf16)w[i];
}

// ------------------------------------------------- K1: WMMA GEMM + bias + elu
// eqk[m, j] = elu( sum_k x[m,k] * W[j,k] + b[j] ) + 1
// Block: 256 threads = 8 waves. Block covers 64 rows x 192 cols.
// Wave (w): m-subtile = w>>1 (4 x 16 rows), n-half = w&1 (6 x 16 cols).
__global__ __launch_bounds__(256) void la_gemm(const float* __restrict__ x,
                                               const __bf16* __restrict__ wb,
                                               const float* __restrict__ bias,
                                               float* __restrict__ eqk) {
    const int lane  = threadIdx.x & 31;
    const int wave  = threadIdx.x >> 5;
    const int mSub  = wave >> 1;
    const int nHalf = wave & 1;
    const int mb    = blockIdx.x * 64 + mSub * 16;
    const int half  = lane >> 4;   // K-half selector
    const int l16   = lane & 15;   // row (A) / col (B/D)

    // A fragments: 16x32 bf16, contiguous K order (matches B):
    //   lane<16: K = {0..15}+c*32 ; lane>=16: K = {16..31}+c*32
    bf16x16 afr[3];
    const float* xrow = x + (size_t)(mb + l16) * CH;
#pragma unroll
    for (int c = 0; c < 3; ++c) {
        const int kk = c * 32 + half * 16;
#pragma unroll
        for (int i = 0; i < 16; ++i) {
            const int k = kk + i;
            afr[c][i] = (__bf16)xrow[k];
        }
    }

#pragma unroll
    for (int t = 0; t < 6; ++t) {
        const int n0 = (nHalf * 6 + t) * 16;
        // B fragment: col = l16, K = c*32 + i + 16*half  (W is N-major: W[j,k])
        const __bf16* wrow = wb + (size_t)(n0 + l16) * CH + half * 16;
        floatx8 acc = {0.f, 0.f, 0.f, 0.f, 0.f, 0.f, 0.f, 0.f};
#pragma unroll
        for (int c = 0; c < 3; ++c) {
            bf16x16 bfr;
#pragma unroll
            for (int i = 0; i < 16; ++i) bfr[i] = wrow[c * 32 + i];
            acc = __builtin_amdgcn_wmma_f32_16x16x32_bf16(
                false, afr[c], false, bfr, (short)0, acc, false, false);
        }
        const float bv = bias[n0 + l16];
        // D layout: vgpr r, lane l -> row = r + 8*(l>>4), col = l&15
#pragma unroll
        for (int r = 0; r < 8; ++r) {
            const int gm = mb + r + half * 8;
            float v = acc[r] + bv;
            v = (v > 0.f) ? (v + 1.f) : __expf(v);   // elu(v)+1
            eqk[(size_t)gm * QKC + n0 + l16] = v;
        }
    }
}

// ------------------------------------------- K2: k column-sum partials (mean)
__global__ __launch_bounds__(96) void la_ksum_part(const float* __restrict__ eqk,
                                                   float* __restrict__ kpart) {
    const int b     = blockIdx.x >> 6;
    const int chunk = blockIdx.x & 63;         // 64 chunks of 256 tokens
    const int c     = threadIdx.x;
    const float* p  = eqk + ((size_t)(b * NTOK + chunk * 256)) * QKC + CH + c;
    float s = 0.f;
    for (int t = 0; t < 256; ++t) s += p[(size_t)t * QKC];
    kpart[(size_t)blockIdx.x * CH + c] = s;
}

__global__ void la_ksum_reduce(const float* __restrict__ kpart,
                               float* __restrict__ kmean) {
    const int tid = blockIdx.x * blockDim.x + threadIdx.x;
    if (tid >= BATCH * CH) return;
    const int b = tid / CH, c = tid % CH;
    float s = 0.f;
    for (int ch = 0; ch < 64; ++ch) s += kpart[(size_t)(b * 64 + ch) * CH + c];
    kmean[tid] = s / (float)NTOK;
}

// ----------------------------- K3: kv = (1/n) sum_n rope(k) outer v, partials
// grid = (16 chunks, 32 (b,h)); 192 threads; thread t owns (e=t%24, d=3*(t/24)+{0,1,2})
__global__ __launch_bounds__(192) void la_kv_part(const float* __restrict__ eqk,
                                                  const float* __restrict__ x,
                                                  float* __restrict__ kvp) {
    __shared__ float kr[8][HD];
    __shared__ float vv[8][HD];
    __shared__ float th[12];
    const int chunk = blockIdx.x;
    const int bh    = blockIdx.y;
    const int b = bh >> 2, h = bh & 3;
    const int t = threadIdx.x;
    if (t < 12) th[t] = __powf(10000.f, -(float)(h * 12 + t) / 48.f);
    const int e  = t % HD;   // channel within head / e-column
    const int dg = t / HD;   // token slot for loads / d-group for accum (0..7)
    float a0 = 0.f, a1 = 0.f, a2 = 0.f;
    const int n0 = chunk * 1024;

    for (int base = 0; base < 1024; base += 8) {
        __syncthreads();                       // prev accum done; th visible
        const int n = n0 + base + dg;
        kr[dg][e] = eqk[((size_t)(b * NTOK) + n) * QKC + CH + h * HD + e];
        vv[dg][e] = x  [((size_t)(b * NTOK) + n) * CH  + h * HD + e];
        __syncthreads();
        if (t < 96) {                          // in-place RoPE on k tile
            const int ltk = t / 12, jj = t % 12;
            const int nn  = n0 + base + ltk;
            const float ang = (float)(nn & (HW - 1)) * th[jj];
            const float sn = __sinf(ang), cs = __cosf(ang);
            const float re = kr[ltk][2 * jj], im = kr[ltk][2 * jj + 1];
            kr[ltk][2 * jj]     = re * cs - im * sn;
            kr[ltk][2 * jj + 1] = re * sn + im * cs;
        }
        __syncthreads();
#pragma unroll
        for (int q = 0; q < 8; ++q) {
            const float ve = vv[q][e];
            a0 += kr[q][3 * dg + 0] * ve;
            a1 += kr[q][3 * dg + 1] * ve;
            a2 += kr[q][3 * dg + 2] * ve;
        }
    }
    const float sc = 1.f / (float)NTOK;        // (k/sqrt(n))*(v/sqrt(n))
    const size_t o = (((size_t)chunk * 32 + bh) * 192 + t) * 3;
    kvp[o + 0] = a0 * sc; kvp[o + 1] = a1 * sc; kvp[o + 2] = a2 * sc;
}

__global__ __launch_bounds__(192) void la_kv_reduce(const float* __restrict__ kvp,
                                                    float* __restrict__ kv) {
    const int bh = blockIdx.x, t = threadIdx.x;
    const int e = t % HD, dg = t / HD;
    float s0 = 0.f, s1 = 0.f, s2 = 0.f;
    for (int ch = 0; ch < 16; ++ch) {
        const size_t o = (((size_t)ch * 32 + bh) * 192 + t) * 3;
        s0 += kvp[o]; s1 += kvp[o + 1]; s2 += kvp[o + 2];
    }
    const size_t kb = (size_t)bh * HD * HD;
    kv[kb + (3 * dg + 0) * HD + e] = s0;
    kv[kb + (3 * dg + 1) * HD + e] = s1;
    kv[kb + (3 * dg + 2) * HD + e] = s2;
}

// --------------------------- K4: out = rope(q)@kv * z + LePE(v) ; one token/blk
__global__ __launch_bounds__(96) void la_out(const float* __restrict__ eqk,
                                             const float* __restrict__ x,
                                             const float* __restrict__ kv,
                                             const float* __restrict__ kmean,
                                             const float* __restrict__ lw,
                                             const float* __restrict__ lb,
                                             float* __restrict__ out) {
    __shared__ float qv[CH], qr[CH], th[48], z4[NH];
    __shared__ float kvs[NH * HD * HD];        // 9216 B
    const int bn = blockIdx.x;
    const int b = bn >> 14;                    // / 16384
    const int n = bn & (NTOK - 1);
    const int c = threadIdx.x;
    if (c < 48) th[c] = __powf(10000.f, -(float)c / 48.f);
    qv[c] = eqk[(size_t)bn * QKC + c];
    for (int i = c; i < NH * HD * HD; i += CH)
        kvs[i] = kv[(size_t)b * NH * HD * HD + i];
    __syncthreads();
    {   // rope on q (pairs (2j,2j+1); angle index = n mod 128)
        const int j = c >> 1;
        const float ang = (float)(n & (HW - 1)) * th[j];
        const float sn = __sinf(ang), cs = __cosf(ang);
        if (c & 1) qr[c] = qv[c - 1] * sn + qv[c] * cs;
        else       qr[c] = qv[c] * cs - qv[c + 1] * sn;
    }
    if (c < NH) {                              // z = 1/(q . k_mean + 1e-6)
        float s = 0.f;
        for (int d = 0; d < HD; ++d) s += qv[c * HD + d] * kmean[b * CH + c * HD + d];
        z4[c] = 1.f / (s + 1e-6f);
    }
    __syncthreads();
    const int h = c / HD, e = c % HD;
    float acc = 0.f;
#pragma unroll
    for (int d = 0; d < HD; ++d) acc += qr[h * HD + d] * kvs[(h * HD + d) * HD + e];
    acc *= z4[h];
    // LePE depthwise 3x3, SAME zero-pad; taps coalesced across channels
    const int y = n >> 7, xx = n & (HW - 1);
    float lp = lb[c];
#pragma unroll
    for (int dy = -1; dy <= 1; ++dy) {
        const int yy = y + dy;
        if (yy < 0 || yy >= HW) continue;
#pragma unroll
        for (int dx = -1; dx <= 1; ++dx) {
            const int x2 = xx + dx;
            if (x2 < 0 || x2 >= HW) continue;
            lp += x[((size_t)b * NTOK + yy * HW + x2) * CH + c] *
                  lw[c * 9 + (dy + 1) * 3 + (dx + 1)];
        }
    }
    out[(size_t)bn * CH + c] = acc + lp;
}

// ------------------------------------------------------------------- launcher
extern "C" void kernel_launch(void* const* d_in, const int* in_sizes, int n_in,
                              void* d_out, int out_size, void* d_ws, size_t ws_size,
                              hipStream_t stream) {
    const float* x    = (const float*)d_in[0];
    const float* qk_w = (const float*)d_in[1];
    const float* qk_b = (const float*)d_in[2];
    const float* lw   = (const float*)d_in[3];
    const float* lb   = (const float*)d_in[4];
    float* out = (float*)d_out;

    char* ws = (char*)d_ws;
    float*  eqk   = (float*)(ws + 0);                     // 100,663,296 B
    __bf16* wb    = (__bf16*)(ws + 100663296);            //      36,864 B
    float*  kpart = (float*)(ws + 100700160);             //     196,608 B
    float*  kmean = (float*)(ws + 100896768);             //       3,072 B
    float*  kvp   = (float*)(ws + 100899840);             //   1,179,648 B
    float*  kv    = (float*)(ws + 102079488);             //      73,728 B

    la_wconv<<<(QKC * CH + 255) / 256, 256, 0, stream>>>(qk_w, wb);
    la_gemm<<<(BATCH * NTOK) / 64, 256, 0, stream>>>(x, wb, qk_b, eqk);
    la_ksum_part<<<BATCH * 64, 96, 0, stream>>>(eqk, kpart);
    la_ksum_reduce<<<3, 256, 0, stream>>>(kpart, kmean);
    la_kv_part<<<dim3(16, 32), 192, 0, stream>>>(eqk, x, kvp);
    la_kv_reduce<<<32, 192, 0, stream>>>(kvp, kv);
    la_out<<<BATCH * NTOK, 96, 0, stream>>>(eqk, x, kv, kmean, lw, lb, out);
}
